// GRVQS_2559800508668
// MI455X (gfx1250) — matrix operand order repaired and checked by reading, not compiled, not run on hardware
//
#include <hip/hip_runtime.h>

// ---------------------------------------------------------------------------
// Problem constants (fixed by the reference: setup_inputs()).
// ---------------------------------------------------------------------------
#define BATCH 4
#define SEQ   2048
#define DIM   1280
#define FFD   5120
#define NHEAD 16
#define HD    80
#define HDP   96            // head dim padded to 3*32 for WMMA K-steps
#define MR    (BATCH*SEQ)   // 8192 rows for all row-major GEMMs
#define NL    2
#define VG    2
#define VQ    8
#define VK    1024
#define VDC   64
#define VDG   640

typedef __attribute__((ext_vector_type(16))) __bf16 v16bf;
typedef __attribute__((ext_vector_type(8)))  __bf16 v8bf;
typedef __attribute__((ext_vector_type(8)))  float  v8f;
typedef int v4i __attribute__((vector_size(16)));   // async-LDS builtin operand

union Frag16 { v16bf v; v8bf h[2]; };

// --- CDNA5 async global->LDS staging (ASYNCcnt path), with safe fallback ----
#if defined(__has_builtin)
#if __has_builtin(__builtin_amdgcn_global_load_async_to_lds_b128)
#define HAVE_ASYNC_LDS 1
#endif
#endif
#ifndef HAVE_ASYNC_LDS
#define HAVE_ASYNC_LDS 0
#endif

typedef __attribute__((address_space(1))) v4i* gv4i_p;   // global
typedef __attribute__((address_space(3))) v4i* lv4i_p;   // LDS

__device__ __forceinline__ void wait_async0() {
#if defined(__has_builtin) && __has_builtin(__builtin_amdgcn_s_wait_asynccnt)
  __builtin_amdgcn_s_wait_asynccnt(0);
#else
  asm volatile("s_wait_asynccnt 0x0" ::: "memory");
#endif
}

__device__ __forceinline__ v8f wmma_bf16(v16bf a, v16bf b, v8f c) {
  // 8 args: (neg_a, A, neg_b, B, c_mod, C, reuse_a, reuse_b)
  return __builtin_amdgcn_wmma_f32_16x16x32_bf16(false, a, false, b,
                                                 (short)0, c, false, false);
}

#define ATP 40   // padded LDS row stride (bf16 elems) to stagger banks

// Stage a 64x32 bf16 A tile (rows mbase..mbase+63, K-cols kk..kk+31) into LDS.
// 128 threads x 2 chunks of 16B each, via GLOBAL_LOAD_ASYNC_TO_LDS_B128.
__device__ __forceinline__ void stage_A(const __bf16* __restrict__ A,
                                        __bf16* dstbase, int mbase, int kk,
                                        int Kd, int tid) {
#pragma unroll
  for (int c = tid; c < 256; c += 128) {
    const int row = c >> 2, part = c & 3;
    const __bf16* g = A + (size_t)(mbase + row) * Kd + kk + part * 8;
    __bf16* l = dstbase + row * ATP + part * 8;
#if HAVE_ASYNC_LDS
    void* gp = (void*)g;   // strip const, then addrspace+pointee cast
    void* lp = (void*)l;
    __builtin_amdgcn_global_load_async_to_lds_b128((gv4i_p)gp, (lv4i_p)lp, 0, 0);
#else
    *(v8bf*)l = *(const v8bf*)g;
#endif
  }
}

// ---------------------------------------------------------------------------
// Generic WMMA GEMM:  C[M,N] = A_bf16[M,K] @ BT_bf16[N,K]^T  (+bias)(+act)(+res)
// Block = 128 threads = 4 waves; block tile 64(M) x 64(N); each wave does a
// 64x16 strip (4 C fragments sharing one B fragment per K-step). The shared
// A tile is staged through LDS (double-buffered, async global->LDS).
// act: 0 = none, 1 = exact GELU.
// ---------------------------------------------------------------------------
__global__ void __launch_bounds__(128)
k_gemm(const __bf16* __restrict__ A, const __bf16* __restrict__ BT,
       const float* __restrict__ bias, const float* __restrict__ res,
       float* __restrict__ C, __bf16* __restrict__ Cbf,
       int Nd, int Kd, int act)
{
  __shared__ __align__(16) __bf16 Atile[2][64 * ATP];
  const int tid  = threadIdx.x;
  const int lane = tid & 31;
  const int wv   = tid >> 5;
  const int hf   = lane >> 4;     // lane half (wave32)
  const int l15  = lane & 15;
  const int mbase = blockIdx.x * 64;
  const int col   = blockIdx.y * 64 + wv * 16 + l15;

  v8f acc[4];
#pragma unroll
  for (int t = 0; t < 4; ++t)
#pragma unroll
    for (int i = 0; i < 8; ++i) acc[t][i] = 0.0f;

  stage_A(A, &Atile[0][0], mbase, 0, Kd, tid);

  const __bf16* brow = BT + (size_t)col * Kd;
  int bufi = 0;
  for (int kk = 0; kk < Kd; kk += 32) {
#if HAVE_ASYNC_LDS
    wait_async0();                       // own wave's async stores done
#endif
    __syncthreads();                     // whole tile visible to all waves
    if (kk + 32 < Kd) stage_A(A, &Atile[bufi ^ 1][0], mbase, kk + 32, Kd, tid);

    // B fragment (32x16): lane half selects contiguous K-half of 16
    Frag16 fb;
    const __bf16* bp = brow + kk + hf * 16;
    fb.h[0] = *(const v8bf*)(bp);
    fb.h[1] = *(const v8bf*)(bp + 8);
    if (kk + 32 < Kd) __builtin_prefetch(brow + kk + 32, 0, 1);

#pragma unroll
    for (int t = 0; t < 4; ++t) {
      // A fragment (16x32) from LDS: elements 0-15 -> K=16*hf+e
      const __bf16* ap = &Atile[bufi][(t * 16 + l15) * ATP + hf * 16];
      Frag16 fa;
      fa.h[0] = *(const v8bf*)(ap);
      fa.h[1] = *(const v8bf*)(ap + 8);
      acc[t] = wmma_bf16(fa.v, fb.v, acc[t]);
    }
    __syncthreads();                     // reads done before buffer reuse
    bufi ^= 1;
  }

  const float bv = bias ? bias[col] : 0.0f;
#pragma unroll
  for (int t = 0; t < 4; ++t)
#pragma unroll
    for (int i = 0; i < 8; ++i) {
      const int row = mbase + t*16 + i + 8*hf;   // C layout: VGPR i, lane half
      const size_t off = (size_t)row * Nd + col;
      float v = acc[t][i] + bv;
      if (act == 1) v = 0.5f * v * (1.0f + erff(v * 0.70710678f));
      if (res) v += res[off];
      if (C)   C[off]   = v;
      if (Cbf) Cbf[off] = (__bf16)v;
    }
}

// ---------------------------------------------------------------------------
// Flash attention: one wave per (b, h, 16-query tile). Q/K padded [.,96] bf16,
// V transposed [b][h][80][SEQ] bf16. Online softmax over 32-key blocks.
// ---------------------------------------------------------------------------
__global__ void __launch_bounds__(128)
k_attn(const __bf16* __restrict__ Qp, const __bf16* __restrict__ Kp,
       const __bf16* __restrict__ VT, float* __restrict__ Out)
{
  __shared__ __bf16 lds_p[4][16 * 32];   // per-wave P tile (16 rows x 32 keys)
  const int lane = threadIdx.x & 31;
  const int wv   = threadIdx.x >> 5;
  const int hf   = lane >> 4;
  const int l15  = lane & 15;

  const int wid = blockIdx.x * 4 + wv;
  const int nq  = SEQ / 16;
  const int qt  = wid % nq;
  const int h   = (wid / nq) % NHEAD;
  const int b   = wid / (nq * NHEAD);

  const __bf16* qb = Qp + ((size_t)(b*NHEAD + h)) * SEQ * HDP;
  const __bf16* kb = Kp + ((size_t)(b*NHEAD + h)) * SEQ * HDP;
  const __bf16* vb = VT + ((size_t)(b*NHEAD + h)) * HD * SEQ;

  // Q fragments: 3 K-steps of 32 covering padded head dim 96
  Frag16 aq[3];
  const int qrow = qt * 16 + l15;
#pragma unroll
  for (int s = 0; s < 3; ++s) {
    const __bf16* p = qb + (size_t)qrow * HDP + s * 32 + hf * 16;
    aq[s].h[0] = *(const v8bf*)(p);
    aq[s].h[1] = *(const v8bf*)(p + 8);
  }

  v8f acc[5];
  float mrow[8], lrow[8];
#pragma unroll
  for (int t = 0; t < 5; ++t)
#pragma unroll
    for (int i = 0; i < 8; ++i) acc[t][i] = 0.0f;
#pragma unroll
  for (int i = 0; i < 8; ++i) { mrow[i] = -3.0e38f; lrow[i] = 0.0f; }

  for (int kbk = 0; kbk < SEQ; kbk += 32) {
    v8f s0, s1;
#pragma unroll
    for (int i = 0; i < 8; ++i) { s0[i] = 0.0f; s1[i] = 0.0f; }
    // S = Q @ K^T for 32 keys (two 16-key column groups)
#pragma unroll
    for (int s = 0; s < 3; ++s) {
      Frag16 fb;
      const __bf16* p0 = kb + (size_t)(kbk + l15) * HDP + s * 32 + hf * 16;
      fb.h[0] = *(const v8bf*)(p0);
      fb.h[1] = *(const v8bf*)(p0 + 8);
      s0 = wmma_bf16(aq[s].v, fb.v, s0);
      const __bf16* p1 = kb + (size_t)(kbk + 16 + l15) * HDP + s * 32 + hf * 16;
      fb.h[0] = *(const v8bf*)(p1);
      fb.h[1] = *(const v8bf*)(p1 + 8);
      s1 = wmma_bf16(aq[s].v, fb.v, s1);
    }
    // Online softmax (row stats reduced over the 16-lane half-groups)
#pragma unroll
    for (int i = 0; i < 8; ++i) {
      float v = fmaxf(s0[i], s1[i]);
      for (int m = 1; m < 16; m <<= 1) v = fmaxf(v, __shfl_xor(v, m, 32));
      const float mn = fmaxf(mrow[i], v);
      const float corr = __expf(mrow[i] - mn);
      mrow[i] = mn;
      lrow[i] *= corr;
#pragma unroll
      for (int t = 0; t < 5; ++t) acc[t][i] *= corr;
      const float p0 = __expf(s0[i] - mn);
      const float p1 = __expf(s1[i] - mn);
      float rs = p0 + p1;
      for (int m = 1; m < 16; m <<= 1) rs += __shfl_xor(rs, m, 32);
      lrow[i] += rs;
      lds_p[wv][(i + 8*hf) * 32 + l15]      = (__bf16)p0;
      lds_p[wv][(i + 8*hf) * 32 + 16 + l15] = (__bf16)p1;
    }
    // Re-layout P as an A fragment via LDS (same-wave LDS is in-order)
    Frag16 pf;
    const __bf16* lp = &lds_p[wv][l15 * 32 + hf * 16];
    pf.h[0] = *(const v8bf*)(lp);
    pf.h[1] = *(const v8bf*)(lp + 8);
    // O += P @ V  (5 tiles of 16 head-dims)
#pragma unroll
    for (int t = 0; t < 5; ++t) {
      Frag16 fv;
      const __bf16* vp = vb + (size_t)(t*16 + l15) * SEQ + kbk + hf * 16;
      fv.h[0] = *(const v8bf*)(vp);
      fv.h[1] = *(const v8bf*)(vp + 8);
      acc[t] = wmma_bf16(pf.v, fv.v, acc[t]);
    }
  }
  // Normalize and write out (f32, [M][DIM])
#pragma unroll
  for (int t = 0; t < 5; ++t)
#pragma unroll
    for (int i = 0; i < 8; ++i) {
      const int row  = qt * 16 + i + 8*hf;
      const int cold = h * HD + t * 16 + l15;
      Out[((size_t)(b*SEQ + row)) * DIM + cold] = acc[t][i] / lrow[i];
    }
}

// ---------------------------------------------------------------------------
// LayerNorm over D=1280, one wave per row; bf16 output for GEMM input.
// ---------------------------------------------------------------------------
__global__ void __launch_bounds__(256)
k_ln(const float* __restrict__ X, const float* __restrict__ S,
     const float* __restrict__ Bb, __bf16* __restrict__ O)
{
  const int lane = threadIdx.x & 31;
  const int row  = blockIdx.x * 8 + (threadIdx.x >> 5);
  const float* x = X + (size_t)row * DIM;
  float s1 = 0.0f, s2 = 0.0f;
  for (int c = lane; c < DIM; c += 32) { const float v = x[c]; s1 += v; s2 += v*v; }
  for (int m = 1; m < 32; m <<= 1) { s1 += __shfl_xor(s1, m, 32); s2 += __shfl_xor(s2, m, 32); }
  const float mean = s1 * (1.0f / DIM);
  const float var  = s2 * (1.0f / DIM) - mean * mean;
  const float inv  = rsqrtf(var + 1e-5f);
  for (int c = lane; c < DIM; c += 32)
    O[(size_t)row * DIM + c] = (__bf16)((x[c] - mean) * inv * S[c] + Bb[c]);
}

// ---------------------------------------------------------------------------
// Elementwise / packing helpers.
// ---------------------------------------------------------------------------
__global__ void k_copy_f32(const float* __restrict__ s, float* __restrict__ d, size_t n) {
  size_t t = (size_t)blockIdx.x * blockDim.x + threadIdx.x;
  if (t < n) d[t] = s[t];
}
__global__ void k_f2bf(const float* __restrict__ s, __bf16* __restrict__ d, size_t n) {
  size_t t = (size_t)blockIdx.x * blockDim.x + threadIdx.x;
  if (t < n) d[t] = (__bf16)s[t];
}
__global__ void k_zero_f32(float* __restrict__ p, size_t n) {
  size_t t = (size_t)blockIdx.x * blockDim.x + threadIdx.x;
  if (t < n) p[t] = 0.0f;
}
// transpose-convert: src f32 [R][C] -> dst bf16 [C][R]
__global__ void k_tconv(const float* __restrict__ src, __bf16* __restrict__ dst,
                        int R, int C) {
  size_t t = (size_t)blockIdx.x * blockDim.x + threadIdx.x;
  if (t < (size_t)R * C) {
    const int r = (int)(t / C), c = (int)(t % C);
    dst[(size_t)c * R + r] = (__bf16)src[t];
  }
}
// q/k packing: f32 [M][DIM] -> bf16 [B][H][SEQ][96] (zero-padded), with scale
__global__ void k_pack_head(const float* __restrict__ src, __bf16* __restrict__ dst,
                            float scale) {
  size_t t = (size_t)blockIdx.x * blockDim.x + threadIdx.x;
  const size_t n = (size_t)BATCH * NHEAD * SEQ * HDP;
  if (t < n) {
    const int c = (int)(t % HDP);
    const int s = (int)((t / HDP) % SEQ);
    const int h = (int)((t / ((size_t)HDP * SEQ)) % NHEAD);
    const int b = (int)(t / ((size_t)HDP * SEQ * NHEAD));
    float v = 0.0f;
    if (c < HD) v = src[((size_t)(b*SEQ + s)) * DIM + h*HD + c] * scale;
    dst[t] = (__bf16)v;
  }
}
// V transpose: f32 [M][DIM] -> bf16 [B][H][80][SEQ]
__global__ void k_pack_vT(const float* __restrict__ src, __bf16* __restrict__ dst) {
  size_t t = (size_t)blockIdx.x * blockDim.x + threadIdx.x;
  const size_t n = (size_t)BATCH * NHEAD * HD * SEQ;
  if (t < n) {
    const int s = (int)(t % SEQ);
    const int c = (int)((t / SEQ) % HD);
    const int h = (int)((t / ((size_t)SEQ * HD)) % NHEAD);
    const int b = (int)(t / ((size_t)SEQ * HD * NHEAD));
    dst[t] = (__bf16)src[((size_t)(b*SEQ + s)) * DIM + h*HD + c];
  }
}
// GRVQ init: residual <- x group slices, qout <- 0
__global__ void k_grvq_init(const float* __restrict__ x, float* __restrict__ res,
                            float* __restrict__ qout) {
  size_t t = (size_t)blockIdx.x * blockDim.x + threadIdx.x;
  const size_t n = (size_t)VG * MR * VDG;
  if (t < n) {
    const int g  = (int)(t / ((size_t)MR * VDG));
    const int r  = (int)((t / VDG) % MR);
    const int dg = (int)(t % VDG);
    res[t]  = x[(size_t)r * DIM + g * VDG + dg];
    qout[t] = 0.0f;
  }
}
// -0.5 * ||codebook row||^2 per (g, qi, k)
__global__ void k_cbnorm(const float* __restrict__ cb, float* __restrict__ nbias) {
  size_t t = (size_t)blockIdx.x * blockDim.x + threadIdx.x;
  if (t < (size_t)VG * VQ * VK) {
    const float* r = cb + t * VDC;
    float s = 0.0f;
    for (int c = 0; c < VDC; ++c) s += r[c] * r[c];
    nbias[t] = -0.5f * s;
  }
}
// per-row argmax of scores (== argmin of distance); writes indices output
__global__ void __launch_bounds__(256)
k_argmax(const float* __restrict__ S, int* __restrict__ idx,
         float* __restrict__ idxout, int g, int qi) {
  const int lane = threadIdx.x & 31;
  const int row  = blockIdx.x * 8 + (threadIdx.x >> 5);
  const float* s = S + (size_t)row * VK;
  float best = -3.0e38f; int bi = 0;
  for (int c = lane; c < VK; c += 32) {
    const float v = s[c];
    if (v > best) { best = v; bi = c; }
  }
  for (int m = 1; m < 32; m <<= 1) {
    const float ov = __shfl_xor(best, m, 32);
    const int   oi = __shfl_xor(bi,   m, 32);
    if (ov > best || (ov == best && oi < bi)) { best = ov; bi = oi; }
  }
  if (lane == 0) {
    idx[row] = bi;
    const int b = row / SEQ, n = row % SEQ;
    idxout[((size_t)(g*BATCH + b) * SEQ + n) * VQ + qi] = (float)bi;
  }
}
// gather quantized rows (bf16) + accumulate commitment loss
__global__ void __launch_bounds__(256)
k_gather_commit(const __bf16* __restrict__ cbbf, const float* __restrict__ cbf,
                const int* __restrict__ idx, const float* __restrict__ z,
                __bf16* __restrict__ qnt, float* __restrict__ slot, float sc) {
  const int lane = threadIdx.x & 31;
  const int row  = blockIdx.x * 8 + (threadIdx.x >> 5);
  const int k = idx[row];
  float ss = 0.0f;
  for (int c = lane; c < VDC; c += 32) {
    qnt[(size_t)row * VDC + c] = cbbf[(size_t)k * VDC + c];
    const float d = cbf[(size_t)k * VDC + c] - z[(size_t)row * VDC + c];
    ss += d * d;
  }
  for (int m = 1; m < 32; m <<= 1) ss += __shfl_xor(ss, m, 32);
  if (lane == 0) atomicAdd(slot, ss * sc);
}
__global__ void k_update(float* __restrict__ qout, float* __restrict__ res,
                         const float* __restrict__ outq, size_t n) {
  size_t t = (size_t)blockIdx.x * blockDim.x + threadIdx.x;
  if (t < n) { qout[t] += outq[t]; res[t] -= outq[t]; }
}
__global__ void k_write_q(const float* __restrict__ qout, float* __restrict__ dq) {
  size_t t = (size_t)blockIdx.x * blockDim.x + threadIdx.x;
  const size_t n = (size_t)VG * MR * VDG;
  if (t < n) {
    const int g  = (int)(t / ((size_t)MR * VDG));
    const int r  = (int)((t / VDG) % MR);
    const int dg = (int)(t % VDG);
    dq[(size_t)r * DIM + g * VDG + dg] = qout[t];
  }
}

// ---------------------------------------------------------------------------
// Host orchestration.
// ---------------------------------------------------------------------------
static inline unsigned blks(size_t n) { return (unsigned)((n + 255) / 256); }

extern "C" void kernel_launch(void* const* d_in, const int* in_sizes, int n_in,
                              void* d_out, int out_size, void* d_ws, size_t ws_size,
                              hipStream_t stream) {
  (void)in_sizes; (void)n_in; (void)out_size; (void)ws_size;
  const float* x_in  = (const float*)d_in[0];
  const float* ln1s  = (const float*)d_in[1];
  const float* ln1b  = (const float*)d_in[2];
  const float* Wq    = (const float*)d_in[3];
  const float* bq    = (const float*)d_in[4];
  const float* Wk    = (const float*)d_in[5];
  const float* bk    = (const float*)d_in[6];
  const float* Wv    = (const float*)d_in[7];
  const float* bv    = (const float*)d_in[8];
  const float* Wo    = (const float*)d_in[9];
  const float* bo    = (const float*)d_in[10];
  const float* ln2s  = (const float*)d_in[11];
  const float* ln2b  = (const float*)d_in[12];
  const float* Wi    = (const float*)d_in[13];
  const float* bi    = (const float*)d_in[14];
  const float* Wf    = (const float*)d_in[15];
  const float* bf_   = (const float*)d_in[16];
  const float* pin   = (const float*)d_in[17];
  const float* pout  = (const float*)d_in[18];
  const float* cbs   = (const float*)d_in[19];

  float* out_q      = (float*)d_out;
  float* out_idx    = out_q + (size_t)MR * DIM;
  float* out_commit = out_idx + (size_t)VG * BATCH * SEQ * VQ;

  // --- workspace bump allocator (~450 MB) ---
  char* w = (char*)d_ws;
  auto take = [&](size_t bytes) -> char* {
    char* p = w; w += (bytes + 255) & ~(size_t)255; return p;
  };
  float*  xbuf  = (float*) take((size_t)MR * DIM * 4);
  __bf16* hln   = (__bf16*)take((size_t)MR * DIM * 2);
  __bf16* wT    = (__bf16*)take((size_t)DIM * FFD * 2);
  float*  gbuf  = (float*) take((size_t)MR * DIM * 4);
  __bf16* abf   = (__bf16*)take((size_t)MR * DIM * 2);
  __bf16* qpk   = (__bf16*)take((size_t)BATCH * NHEAD * SEQ * HDP * 2);
  __bf16* kpk   = (__bf16*)take((size_t)BATCH * NHEAD * SEQ * HDP * 2);
  __bf16* vtp   = (__bf16*)take((size_t)BATCH * NHEAD * HD * SEQ * 2);
  __bf16* ff1   = (__bf16*)take((size_t)MR * FFD * 2);
  float*  resb  = (float*) take((size_t)VG * MR * VDG * 4);
  float*  qacc  = (float*) take((size_t)VG * MR * VDG * 4);
  __bf16* resbf = (__bf16*)take((size_t)MR * VDG * 2);
  float*  zf    = (float*) take((size_t)MR * VDC * 4);
  __bf16* zbf   = (__bf16*)take((size_t)MR * VDC * 2);
  float*  sbuf  = (float*) take((size_t)MR * VK * 4);
  int*    idxb  = (int*)   take((size_t)MR * 4);
  __bf16* qnt   = (__bf16*)take((size_t)MR * VDC * 2);
  float*  outq  = (float*) take((size_t)MR * VDG * 4);
  __bf16* cbbf  = (__bf16*)take((size_t)VG * VQ * VK * VDC * 2);
  float*  nbias = (float*) take((size_t)VG * VQ * VK * 4);
  __bf16* pT    = (__bf16*)take((size_t)VDC * VDG * 2);

  const dim3 blk256(256), blk128(128);
  const dim3 lnGrid(MR / 8);

  // init: x copy, commit zero, codebook bf16 + norms (once per launch)
  k_copy_f32<<<blks((size_t)MR*DIM), blk256, 0, stream>>>(x_in, xbuf, (size_t)MR*DIM);
  k_zero_f32<<<1, 64, 0, stream>>>(out_commit, (size_t)VG*VQ);
  k_f2bf<<<blks((size_t)VG*VQ*VK*VDC), blk256, 0, stream>>>(cbs, cbbf, (size_t)VG*VQ*VK*VDC);
  k_cbnorm<<<blks((size_t)VG*VQ*VK), blk256, 0, stream>>>(cbs, nbias);

  const float qscale = 0.11180339887498949f; // 80^-0.5

  for (int i = 0; i < NL; ++i) {
    const size_t wdd = (size_t)i * DIM * DIM;
    // ---- attention block ----
    k_ln<<<lnGrid, blk256, 0, stream>>>(xbuf, ln1s + i*DIM, ln1b + i*DIM, hln);
    // Q
    k_tconv<<<blks((size_t)DIM*DIM), blk256, 0, stream>>>(Wq + wdd, wT, DIM, DIM);
    k_gemm<<<dim3(MR/64, DIM/64), blk128, 0, stream>>>(hln, wT, bq + i*DIM, nullptr,
                                                       gbuf, nullptr, DIM, DIM, 0);
    k_pack_head<<<blks((size_t)BATCH*NHEAD*SEQ*HDP), blk256, 0, stream>>>(gbuf, qpk, qscale);
    // K
    k_tconv<<<blks((size_t)DIM*DIM), blk256, 0, stream>>>(Wk + wdd, wT, DIM, DIM);
    k_gemm<<<dim3(MR/64, DIM/64), blk128, 0, stream>>>(hln, wT, bk + i*DIM, nullptr,
                                                       gbuf, nullptr, DIM, DIM, 0);
    k_pack_head<<<blks((size_t)BATCH*NHEAD*SEQ*HDP), blk256, 0, stream>>>(gbuf, kpk, 1.0f);
    // V
    k_tconv<<<blks((size_t)DIM*DIM), blk256, 0, stream>>>(Wv + wdd, wT, DIM, DIM);
    k_gemm<<<dim3(MR/64, DIM/64), blk128, 0, stream>>>(hln, wT, bv + i*DIM, nullptr,
                                                       gbuf, nullptr, DIM, DIM, 0);
    k_pack_vT<<<blks((size_t)BATCH*NHEAD*HD*SEQ), blk256, 0, stream>>>(gbuf, vtp);
    // flash attention -> gbuf (f32 [M][DIM])
    k_attn<<<dim3(BATCH*NHEAD*(SEQ/16)/4), blk128, 0, stream>>>(qpk, kpk, vtp, gbuf);
    // output projection + residual (x <- x + attn @ Wo + bo)
    k_f2bf<<<blks((size_t)MR*DIM), blk256, 0, stream>>>(gbuf, abf, (size_t)MR*DIM);
    k_tconv<<<blks((size_t)DIM*DIM), blk256, 0, stream>>>(Wo + wdd, wT, DIM, DIM);
    k_gemm<<<dim3(MR/64, DIM/64), blk128, 0, stream>>>(abf, wT, bo + i*DIM, xbuf,
                                                       xbuf, nullptr, DIM, DIM, 0);
    // ---- FFN block ----
    k_ln<<<lnGrid, blk256, 0, stream>>>(xbuf, ln2s + i*DIM, ln2b + i*DIM, hln);
    k_tconv<<<blks((size_t)DIM*FFD), blk256, 0, stream>>>(Wi + (size_t)i*DIM*FFD, wT, DIM, FFD);
    k_gemm<<<dim3(MR/64, FFD/64), blk128, 0, stream>>>(hln, wT, bi + i*FFD, nullptr,
                                                       nullptr, ff1, FFD, DIM, 1);
    k_tconv<<<blks((size_t)FFD*DIM), blk256, 0, stream>>>(Wf + (size_t)i*FFD*DIM, wT, FFD, DIM);
    k_gemm<<<dim3(MR/64, DIM/64), blk128, 0, stream>>>(ff1, wT, bf_ + i*DIM, xbuf,
                                                       xbuf, nullptr, DIM, FFD, 0);
  }

  // ---- Grouped residual VQ ----
  k_grvq_init<<<blks((size_t)VG*MR*VDG), blk256, 0, stream>>>(xbuf, resb, qacc);
  for (int g = 0; g < VG; ++g) {
    for (int qi = 0; qi < VQ; ++qi) {
      const size_t gq = (size_t)g * VQ + qi;
      float* resg = resb + (size_t)g * MR * VDG;
      float* qag  = qacc + (size_t)g * MR * VDG;
      // z = residual @ proj_in
      k_f2bf<<<blks((size_t)MR*VDG), blk256, 0, stream>>>(resg, resbf, (size_t)MR*VDG);
      k_tconv<<<blks((size_t)VDG*VDC), blk256, 0, stream>>>(pin + gq*VDG*VDC, pT, VDG, VDC);
      k_gemm<<<dim3(MR/64, VDC/64), blk128, 0, stream>>>(resbf, pT, nullptr, nullptr,
                                                         zf, zbf, VDC, VDG, 0);
      // scores = z @ cb^T - 0.5*||cb||^2 ; argmin dist == argmax score
      k_gemm<<<dim3(MR/64, VK/64), blk128, 0, stream>>>(zbf, cbbf + gq*VK*VDC,
                                                        nbias + gq*VK, nullptr,
                                                        sbuf, nullptr, VK, VDC, 0);
      k_argmax<<<dim3(MR/8), blk256, 0, stream>>>(sbuf, idxb, out_idx, g, qi);
      k_gather_commit<<<dim3(MR/8), blk256, 0, stream>>>(cbbf + gq*VK*VDC,
                                                         cbs + gq*VK*VDC, idxb, zf, qnt,
                                                         out_commit + gq,
                                                         1.0f / ((float)MR * VDC));
      // out = quant @ proj_out ; qout += out ; residual -= out
      k_tconv<<<blks((size_t)VDC*VDG), blk256, 0, stream>>>(pout + gq*VDC*VDG, pT, VDC, VDG);
      k_gemm<<<dim3(MR/64, VDG/64), blk128, 0, stream>>>(qnt, pT, nullptr, nullptr,
                                                         outq, nullptr, VDG, VDC, 0);
      k_update<<<blks((size_t)MR*VDG), blk256, 0, stream>>>(qag, resg, outq, (size_t)MR*VDG);
    }
  }
  k_write_q<<<blks((size_t)VG*MR*VDG), blk256, 0, stream>>>(qacc, out_q);
}
